// STFNet_70901320122647
// MI455X (gfx1250) — matrix operand order repaired and checked by reading, not compiled, not run on hardware
//
#include <hip/hip_runtime.h>
#include <hip/hip_bf16.h>
#include <math.h>

// ---------------- CDNA5 WMMA types ----------------
typedef __attribute__((ext_vector_type(16))) __bf16 v16bf;
typedef __attribute__((ext_vector_type(8)))  float  v8f;
typedef __attribute__((ext_vector_type(4)))  int    v4i_;

union FragU { uint4 q[2]; v16bf v; };

#define N_ROWS  65536
#define HIDDEN  512
#define SDIM    64     // STATE_DIM
#define D2      66     // OUT_DIM^2 + 2
#define OUTP    80     // D2 padded to multiple of 16
#define ROWS_WG 64     // rows per workgroup in MLP kernel

#if defined(__HIP_DEVICE_COMPILE__) && __has_builtin(__builtin_amdgcn_global_load_async_to_lds_b128)
#define HAVE_ASYNC_LDS 1
#else
#define HAVE_ASYNC_LDS 0
#endif

// f32 -> bf16 round-to-nearest-even
__device__ __forceinline__ unsigned short f2bf(float f) {
  unsigned int u = __float_as_uint(f);
  u += 0x7FFFu + ((u >> 16) & 1u);
  return (unsigned short)(u >> 16);
}

// A-fragment (16x32 bf16, MxK) per ISA 7.12.2:
// lane<16: row=lane,   K = k0+[0..7] then k0+[16..23]
// lane>=16: row=lane-16, K = k0+[8..15] then k0+[24..31]
__device__ __forceinline__ v16bf load_a_frag(const unsigned short* base, int pitch,
                                             int r0, int k0, int lane) {
  FragU f;
  const int lr = lane & 15;
  const int lh = lane >> 4;
  const unsigned short* p = base + (r0 + lr) * pitch + k0 + lh * 8;
  f.q[0] = *(const uint4*)(p);
  f.q[1] = *(const uint4*)(p + 16);
  return f.v;
}

// B-fragment (32x16 bf16, KxN) from TRANSPOSED weight Wt[n][k] (pitch = kdim):
// lane<16: col=lane,   K = k0+[0..7] then k0+[16..23]
// lane>=16: col=lane-16, K = k0+[8..15] then k0+[24..31]
__device__ __forceinline__ v16bf load_b_frag(const unsigned short* Wt, int kdim,
                                             int n0, int k0, int lane) {
  FragU f;
  const int lr = lane & 15;
  const int lh = lane >> 4;
  const unsigned short* p = Wt + (n0 + lr) * kdim + k0 + lh * 8;
  f.q[0] = *(const uint4*)(p);
  f.q[1] = *(const uint4*)(p + 16);
  return f.v;
}

__device__ __forceinline__ v8f wmma_bf16(v16bf a, v16bf b, v8f c) {
  // (neg_a, A, neg_b, B, c_mod, C, reuse_a, reuse_b)
  return __builtin_amdgcn_wmma_f32_16x16x32_bf16(false, a, false, b, (short)0, c,
                                                 false, false);
}

// ---------------- weight prep: transpose + convert to bf16 ----------------
// W1t: 512x64, W2t: 512x512, W3t: 80x512 (zero-padded cols 66..79)
__global__ __launch_bounds__(256)
void prep_weights(const float* __restrict__ W1, const float* __restrict__ W2,
                  const float* __restrict__ W3,
                  unsigned short* __restrict__ W1t,
                  unsigned short* __restrict__ W2t,
                  unsigned short* __restrict__ W3t) {
  int i = blockIdx.x * blockDim.x + threadIdx.x;
  const int n1 = HIDDEN * SDIM;          // 32768
  const int n2 = HIDDEN * HIDDEN;        // 262144
  const int n3 = OUTP * HIDDEN;          // 40960
  if (i < n1) {
    int n = i / SDIM, k = i % SDIM;
    W1t[i] = f2bf(W1[k * HIDDEN + n]);
  } else if (i < n1 + n2) {
    int j = i - n1;
    int n = j / HIDDEN, k = j % HIDDEN;
    W2t[j] = f2bf(W2[k * HIDDEN + n]);
  } else if (i < n1 + n2 + n3) {
    int j = i - n1 - n2;
    int n = j / HIDDEN, k = j % HIDDEN;
    W3t[j] = (n < D2) ? f2bf(W3[k * D2 + n]) : (unsigned short)0;
  }
}

// ---------------- fused MLP (WMMA bf16, f32 accum) ----------------
// grid = N_ROWS/ROWS_WG = 1024, block = 128 (4 waves); wave w owns rows [w*16, w*16+16)
// LDS: zA 64x64 bf16 (8KB) | h1 64x512 bf16 (64KB) | h2 64x512 bf16 (64KB) = 136KB
// -> 2 workgroups per WGP within the 320KB LDS budget.
__global__ __launch_bounds__(128)
void mlp_fused(const float* __restrict__ Z,
               const unsigned short* __restrict__ W1t,
               const unsigned short* __restrict__ W2t,
               const unsigned short* __restrict__ W3t,
               const float* __restrict__ b1, const float* __restrict__ b2,
               const float* __restrict__ b3,
               float* __restrict__ T_out,     // N x 64
               float* __restrict__ par_out) { // N x 2  (masked exp applied)
  extern __shared__ unsigned short lds[];
  unsigned short* zA = lds;                       // ROWS_WG*SDIM
  unsigned short* h1 = zA + ROWS_WG * SDIM;       // ROWS_WG*HIDDEN
  unsigned short* h2 = h1 + ROWS_WG * HIDDEN;     // ROWS_WG*HIDDEN

  const int tid  = threadIdx.x;
  const int lane = tid & 31;
  const int wave = tid >> 5;                      // 0..3
  const int row0 = blockIdx.x * ROWS_WG;
  const int lr   = lane & 15;
  const int lh   = lane >> 4;
  const int m0   = wave * 16;

  // stage 0: Z tile -> bf16 LDS (row-major, pitch SDIM)
#if HAVE_ASYNC_LDS
  {
    // DMA the raw f32 tile into LDS (staged in h2, which is dead until layer 2),
    // then convert LDS->LDS. Tracked by ASYNCcnt.
    typedef __attribute__((address_space(1))) v4i_ GV4;   // global int4
    typedef __attribute__((address_space(3))) v4i_ LV4;   // LDS int4
    float* zstage = (float*)h2;                   // 16KB staging inside 64KB h2
    const float* gz = Z + (size_t)row0 * SDIM;
    for (int i = tid; i < ROWS_WG * SDIM / 4; i += 128) {
      unsigned loff = (unsigned)((char*)(zstage + i * 4) - (char*)lds);
      __builtin_amdgcn_global_load_async_to_lds_b128(
          (GV4*)(__SIZE_TYPE__)(gz + i * 4), (LV4*)loff, 0, 0);
    }
#if __has_builtin(__builtin_amdgcn_s_wait_asynccnt)
    __builtin_amdgcn_s_wait_asynccnt(0);
#else
    asm volatile("s_wait_asynccnt 0x0" ::: "memory");
#endif
    __syncthreads();
    for (int i = tid; i < ROWS_WG * SDIM / 4; i += 128) {
      float4 z4 = *(const float4*)(zstage + i * 4);
      unsigned short* d = &zA[i * 4];
      d[0] = f2bf(z4.x); d[1] = f2bf(z4.y); d[2] = f2bf(z4.z); d[3] = f2bf(z4.w);
    }
  }
#else
  {
    const float4* Zt = (const float4*)(Z + (size_t)row0 * SDIM);
    for (int i = tid; i < ROWS_WG * SDIM / 4; i += 128) {
      float4 z4 = Zt[i];
      unsigned short* d = &zA[i * 4];
      d[0] = f2bf(z4.x); d[1] = f2bf(z4.y); d[2] = f2bf(z4.z); d[3] = f2bf(z4.w);
    }
  }
#endif
  __syncthreads();

  // ---- Layer 1: h1 = relu(Z @ W1 + b1), K = 64 ----
  {
    v16bf a0 = load_a_frag(zA, SDIM, m0, 0,  lane);
    v16bf a1 = load_a_frag(zA, SDIM, m0, 32, lane);
    for (int nt = 0; nt < HIDDEN / 16; ++nt) {
      const int n0 = nt * 16;
      v8f acc = {0.f, 0.f, 0.f, 0.f, 0.f, 0.f, 0.f, 0.f};
      acc = wmma_bf16(a0, load_b_frag(W1t, SDIM, n0, 0,  lane), acc);
      acc = wmma_bf16(a1, load_b_frag(W1t, SDIM, n0, 32, lane), acc);
      const int col  = n0 + lr;
      const float bb = b1[col];
#pragma unroll
      for (int r = 0; r < 8; ++r) {
        float v = acc[r] + bb;
        v = v > 0.f ? v : 0.f;
        h1[(m0 + r + lh * 8) * HIDDEN + col] = f2bf(v);
      }
    }
  }
  __syncthreads();

  // ---- Layer 2: h2 = relu(h1 @ W2 + b2), K = 512 ----
  // 4-wide N-blocking: one A-fragment feeds 4 WMMAs (32 accum VGPRs live).
  for (int ntb = 0; ntb < HIDDEN / 16; ntb += 4) {
    v8f acc[4];
#pragma unroll
    for (int j = 0; j < 4; ++j) acc[j] = (v8f){0.f,0.f,0.f,0.f,0.f,0.f,0.f,0.f};
#pragma unroll 2
    for (int kk = 0; kk < HIDDEN; kk += 32) {
      __builtin_prefetch((const void*)(W2t + (ntb * 16 + lr) * HIDDEN + kk + 64), 0, 1);
      v16bf af = load_a_frag(h1, HIDDEN, m0, kk, lane);
#pragma unroll
      for (int j = 0; j < 4; ++j)
        acc[j] = wmma_bf16(af, load_b_frag(W2t, HIDDEN, (ntb + j) * 16, kk, lane), acc[j]);
    }
#pragma unroll
    for (int j = 0; j < 4; ++j) {
      const int col  = (ntb + j) * 16 + lr;
      const float bb = b2[col];
#pragma unroll
      for (int r = 0; r < 8; ++r) {
        float v = acc[j][r] + bb;
        v = v > 0.f ? v : 0.f;
        h2[(m0 + r + lh * 8) * HIDDEN + col] = f2bf(v);
      }
    }
  }
  __syncthreads();

  // ---- Layer 3: out = h2 @ W3 + b3 -> T (cols 0..63), params=exp (64,65) ----
  // 5 N-tiles, all blocked on one A-fragment per K-step.
  {
    v8f acc[5];
#pragma unroll
    for (int j = 0; j < 5; ++j) acc[j] = (v8f){0.f,0.f,0.f,0.f,0.f,0.f,0.f,0.f};
#pragma unroll 2
    for (int kk = 0; kk < HIDDEN; kk += 32) {
      v16bf af = load_a_frag(h2, HIDDEN, m0, kk, lane);
#pragma unroll
      for (int j = 0; j < 5; ++j)
        acc[j] = wmma_bf16(af, load_b_frag(W3t, HIDDEN, j * 16, kk, lane), acc[j]);
    }
#pragma unroll
    for (int j = 0; j < 5; ++j) {
      const int col  = j * 16 + lr;
      const float bb = (col < D2) ? b3[col] : 0.f;
#pragma unroll
      for (int r = 0; r < 8; ++r) {
        const int grow = row0 + m0 + r + lh * 8;
        float v = acc[j][r] + bb;
        if (col < 64) {
          T_out[(size_t)grow * 64 + col] = v;
        } else if (col < D2) {
          // params = exp(out[:,-2:]) * row_mask (mask folded in here, exact 0)
          float p = (grow < D2) ? expf(v) : 0.f;
          par_out[(size_t)grow * 2 + (col - 64)] = p;
        }
      }
    }
  }
}

// ---------------- sequential Kalman scan (latency-bound) ----------------
// ONE wave32 workgroup: barrier ops lower to S_NOP (ISA 3.1, single-wave WG).
// Lane t owns matrix elements t and t+32 (rows r0=t>>3 and r0+4, column t&7).
// Q_kf = Hm (params[:,0]), H_kf = Qm (params[:,1]) -- note the swap in reference.
// Fast path: row_mask makes Q=H=0 for all steps beyond the first 65, where
// K = P_pred inv(P_pred) = I exactly => a_upd = y, P_upd = I - P_pred.
__global__ __launch_bounds__(32)
void kf_scan(const float* __restrict__ Y,     // N x 8
             const float* __restrict__ Tm,    // N x 64
             const float* __restrict__ Pr,    // N x 2 (masked exp params)
             float* __restrict__ alpha_out,   // N x 8
             float* __restrict__ P_out) {     // N x 64
  __shared__ float T[64], P[64], tmp[64], Pp_[64], Km[64];
  __shared__ float S[8][17];   // augmented [S | I], padded
  __shared__ float colp[8], a[8], apred[8], y[8];

  const int t  = threadIdx.x;        // 0..31
  const int r0 = t >> 3, c0 = t & 7; // element t
  const int r1 = r0 + 4;             // element t+32 (same column)

  if (t < 8) { a[t] = 1.f; alpha_out[t] = 1.f; }
  {
    float p0 = (r0 == c0) ? 1.f : 0.f;
    float p1 = (r1 == c0) ? 1.f : 0.f;
    P[t] = p0; P[t + 32] = p1;
    P_out[t] = p0; P_out[t + 32] = p1;
  }
  __syncthreads();

  for (int i = 0; i < N_ROWS - 1; ++i) {
    T[t]      = Tm[(size_t)i * 64 + t];
    T[t + 32] = Tm[(size_t)i * 64 + t + 32];
    if (t < 8) y[t] = Y[(size_t)(i + 1) * 8 + t];
    __syncthreads();

    // force scalar (uniform) step parameters -> uniform branch below
    const float qd = __uint_as_float(
        __builtin_amdgcn_readfirstlane(__float_as_uint(Pr[(size_t)(i + 1) * 2 + 0])));
    const float hd = __uint_as_float(
        __builtin_amdgcn_readfirstlane(__float_as_uint(Pr[(size_t)(i + 1) * 2 + 1])));

    if (t < 8) {                     // a_pred = T @ a
      float s = 0.f;
      for (int k = 0; k < 8; ++k) s += T[t * 8 + k] * a[k];
      apred[t] = s;
    }
    {                                // tmp = T @ P
      float s0 = 0.f, s1 = 0.f;
      for (int k = 0; k < 8; ++k) {
        s0 += T[r0 * 8 + k] * P[k * 8 + c0];
        s1 += T[r1 * 8 + k] * P[k * 8 + c0];
      }
      tmp[t] = s0; tmp[t + 32] = s1;
    }
    __syncthreads();
    float pp0, pp1;
    {                                // P_pred = tmp @ T^T + Q
      pp0 = (r0 == c0) ? qd : 0.f;
      pp1 = (r1 == c0) ? qd : 0.f;
      for (int k = 0; k < 8; ++k) {
        pp0 += tmp[r0 * 8 + k] * T[c0 * 8 + k];
        pp1 += tmp[r1 * 8 + k] * T[c0 * 8 + k];
      }
      Pp_[t] = pp0; Pp_[t + 32] = pp1;
    }
    __syncthreads();

    if (qd == 0.f && hd == 0.f) {
      // ---- fast path (99.9% of steps) ----
      if (t < 8) { a[t] = y[t]; alpha_out[(size_t)(i + 1) * 8 + t] = y[t]; }
      const float s0 = ((r0 == c0) ? 1.f : 0.f) - pp0;
      const float s1 = ((r1 == c0) ? 1.f : 0.f) - pp1;
      P[t] = s0; P[t + 32] = s1;
      P_out[(size_t)(i + 1) * 64 + t]      = s0;
      P_out[(size_t)(i + 1) * 64 + t + 32] = s1;
      __syncthreads();
    } else {
      // ---- full Kalman update with 8x8 Gauss-Jordan inverse ----
      S[r0][c0] = pp0 + ((r0 == c0) ? hd : 0.f);
      S[r1][c0] = pp1 + ((r1 == c0) ? hd : 0.f);
      S[r0][c0 + 8] = (r0 == c0) ? 1.f : 0.f;
      S[r1][c0 + 8] = (r1 == c0) ? 1.f : 0.f;
      __syncthreads();
      for (int p = 0; p < 8; ++p) {
        if (c0 == 0) { colp[r0] = S[r0][p]; colp[r1] = S[r1][p]; }
        __syncthreads();
        const float pinv = 1.f / colp[p];
        if (r0 == p) { S[p][c0] *= pinv; S[p][c0 + 8] *= pinv; }
        if (r1 == p) { S[p][c0] *= pinv; S[p][c0 + 8] *= pinv; }
        __syncthreads();
        if (r0 != p) { const float f = colp[r0]; S[r0][c0] -= f * S[p][c0]; S[r0][c0 + 8] -= f * S[p][c0 + 8]; }
        if (r1 != p) { const float f = colp[r1]; S[r1][c0] -= f * S[p][c0]; S[r1][c0 + 8] -= f * S[p][c0 + 8]; }
        __syncthreads();
      }
      {                              // K = P_pred @ inv(S)
        float s0 = 0.f, s1 = 0.f;
        for (int k = 0; k < 8; ++k) {
          s0 += Pp_[r0 * 8 + k] * S[k][c0 + 8];
          s1 += Pp_[r1 * 8 + k] * S[k][c0 + 8];
        }
        Km[t] = s0; Km[t + 32] = s1;
      }
      __syncthreads();
      if (t < 8) {                   // a_upd = a_pred + K (y - a_pred)
        float s = apred[t];
        for (int k = 0; k < 8; ++k) s += Km[t * 8 + k] * (y[k] - apred[k]);
        a[t] = s;
        alpha_out[(size_t)(i + 1) * 8 + t] = s;
      }
      {                              // P_upd = I - K @ P_pred
        float s0 = (r0 == c0) ? 1.f : 0.f;
        float s1 = (r1 == c0) ? 1.f : 0.f;
        for (int k = 0; k < 8; ++k) {
          s0 -= Km[r0 * 8 + k] * Pp_[k * 8 + c0];
          s1 -= Km[r1 * 8 + k] * Pp_[k * 8 + c0];
        }
        __syncthreads();
        P[t] = s0; P[t + 32] = s1;
        P_out[(size_t)(i + 1) * 64 + t]      = s0;
        P_out[(size_t)(i + 1) * 64 + t + 32] = s1;
      }
      __syncthreads();
    }
  }
}

// ---------------- host launcher ----------------
extern "C" void kernel_launch(void* const* d_in, const int* in_sizes, int n_in,
                              void* d_out, int out_size, void* d_ws, size_t ws_size,
                              hipStream_t stream) {
  const float* Y  = (const float*)d_in[0];
  const float* Z  = (const float*)d_in[1];
  const float* W1 = (const float*)d_in[2];
  const float* b1 = (const float*)d_in[3];
  const float* W2 = (const float*)d_in[4];
  const float* b2 = (const float*)d_in[5];
  const float* W3 = (const float*)d_in[6];
  const float* b3 = (const float*)d_in[7];

  // workspace layout (~18 MB total)
  unsigned short* W1t = (unsigned short*)d_ws;               // 512*64
  unsigned short* W2t = W1t + HIDDEN * SDIM;                 // 512*512
  unsigned short* W3t = W2t + HIDDEN * HIDDEN;               // 80*512
  size_t woff = (size_t)(HIDDEN * SDIM + HIDDEN * HIDDEN + OUTP * HIDDEN) * 2;
  woff = (woff + 255) & ~(size_t)255;
  float* T_ws   = (float*)((char*)d_ws + woff);              // N*64
  float* par_ws = T_ws + (size_t)N_ROWS * 64;                // N*2

  float* alpha = (float*)d_out;                              // N*8
  float* Pout  = alpha + (size_t)N_ROWS * 8;                 // N*64

  const int prep_total = HIDDEN * SDIM + HIDDEN * HIDDEN + OUTP * HIDDEN;
  prep_weights<<<(prep_total + 255) / 256, 256, 0, stream>>>(W1, W2, W3, W1t, W2t, W3t);

  const size_t lds_bytes =
      (size_t)(ROWS_WG * SDIM + 2 * ROWS_WG * HIDDEN) * sizeof(unsigned short); // 136KB
  mlp_fused<<<N_ROWS / ROWS_WG, 128, lds_bytes, stream>>>(
      Z, W1t, W2t, W3t, b1, b2, b3, T_ws, par_ws);

  kf_scan<<<1, 32, 0, stream>>>(Y, T_ws, par_ws, alpha, Pout);
}
